// GemmaAttention_7026566497100
// MI455X (gfx1250) — compile-verified, matrix operand order checked by reading, not run on hardware
//
#include <hip/hip_runtime.h>
#include <hip/hip_bf16.h>

// ---------------------------------------------------------------------------
// Gemma attention layer for gfx1250 (MI455X), mixed precision f16-WMMA path.
//   B=2, S=2048, HID=1024, NH=4, HD=256, SCALE=HD^-0.5=0.0625
// ---------------------------------------------------------------------------

typedef __attribute__((ext_vector_type(16))) _Float16 v16h;
typedef __attribute__((ext_vector_type(8)))  _Float16 v8h;
typedef __attribute__((ext_vector_type(8)))  float    v8f;

#define B_    2
#define S_    2048
#define HID_  1024
#define NH_   4
#define HD_   256
#define SCALE_ 0.0625f

// --------- fragment loader (CDNA5 WMMA 16-bit layouts, ISA 7.12.2) ---------
// A-fragment / "row-gather B" : lane idx (0..15) = row, halves = 16 K values:
//   lanes 0-15 : K = k0+0..7   and k0+16..23
//   lanes 16-31: K = k0+8..15  and k0+24..31
// Two 16-byte aligned loads -> global_load_b128 / ds_load_b128.
__device__ __forceinline__ v16h load_frag_row(const _Float16* __restrict__ p,
                                              int ld, int row0, int k0, int lane) {
    const int grp = (lane >> 4) & 1;
    const int idx = lane & 15;
    const _Float16* b = p + (size_t)(row0 + idx) * ld + k0 + grp * 8;
    v8h lo = *(const v8h*)(b);        // K = grp*8 + 0..7
    v8h hi = *(const v8h*)(b + 16);   // K = grp*8 + 16..23
    v16h r;
#pragma unroll
    for (int h = 0; h < 8; ++h) { r[h] = lo[h]; r[h + 8] = hi[h]; }
    return r;
}

__device__ __forceinline__ v8f wmma16x16x32(v16h a, v16h b, v8f c) {
    return __builtin_amdgcn_wmma_f32_16x16x32_f16(false, a, false, b,
                                                  (short)0, c, false, false);
}

// ----------------------------- f32 -> f16 ----------------------------------
__global__ void __launch_bounds__(256) cvt_f32_f16(const float* __restrict__ in,
                                                   _Float16* __restrict__ out,
                                                   int n) {
    int i = blockIdx.x * 256 + threadIdx.x;
    if (i < n) out[i] = (_Float16)in[i];
}

// ------------------- GEMM:  C[M,N](f32) = A[M,K] * W[N,K]^T ----------------
// wave tile: 16(M) x 64(N); 4 WMMA accumulators; A fragment reused 4x.
__global__ void __launch_bounds__(256) gemm_wmma(const _Float16* __restrict__ A,
                                                 const _Float16* __restrict__ W,
                                                 float* __restrict__ C,
                                                 int M, int N, int K) {
    const int lane = threadIdx.x & 31;
    const int wid  = (blockIdx.x * 256 + threadIdx.x) >> 5;
    const int tilesN = N >> 6;
    const int mt = wid / tilesN;
    const int nt = wid - mt * tilesN;
    const int m0 = mt * 16;
    const int n0 = nt * 64;
    if (m0 >= M) return;
    const int idx = lane & 15;

    v8f c0 = {}, c1 = {}, c2 = {}, c3 = {};
    for (int k0 = 0; k0 < K; k0 += 32) {
        if (k0 + 64 < K) {  // prefetch next-next K tile (global_prefetch_b8)
            __builtin_prefetch(A + (size_t)(m0 + idx) * K + k0 + 64, 0, 3);
            __builtin_prefetch(W + (size_t)(n0 + idx) * K + k0 + 64, 0, 3);
        }
        v16h a  = load_frag_row(A, K, m0, k0, lane);
        v16h b0 = load_frag_row(W, K, n0 +  0, k0, lane);
        v16h b1 = load_frag_row(W, K, n0 + 16, k0, lane);
        v16h b2 = load_frag_row(W, K, n0 + 32, k0, lane);
        v16h b3 = load_frag_row(W, K, n0 + 48, k0, lane);
        c0 = wmma16x16x32(a, b0, c0);
        c1 = wmma16x16x32(a, b1, c1);
        c2 = wmma16x16x32(a, b2, c2);
        c3 = wmma16x16x32(a, b3, c3);
    }
    const int grp = (lane >> 4) & 1;
#pragma unroll
    for (int r = 0; r < 8; ++r) {
        const size_t row = (size_t)(m0 + r + grp * 8) * N + n0 + idx;
        C[row +  0] = c0[r];
        C[row + 16] = c1[r];
        C[row + 32] = c2[r];
        C[row + 48] = c3[r];
    }
}

// ----------------- RoPE + scale(Q) + repack to f16 per-head ----------------
// qkv f32 [B,S,3,NH,HD]  ->  q/k/v f16 [B,NH,S,HD];  q *= SCALE
__global__ void __launch_bounds__(256) rope_cvt(const float* __restrict__ qkv,
                                                const float* __restrict__ fre,
                                                const float* __restrict__ fim,
                                                _Float16* __restrict__ qf,
                                                _Float16* __restrict__ kf,
                                                _Float16* __restrict__ vf) {
    const int tid = blockIdx.x * 256 + threadIdx.x;     // B*S*NH*(HD/2) threads
    const int d2 = tid & 127;
    const int h  = (tid >> 7) & 3;
    const int s  = (tid >> 9) & 2047;
    const int b  = tid >> 20;

    const size_t src = (size_t)(b * S_ + s) * (3 * HID_) + h * HD_;
    const size_t dst = ((size_t)(b * NH_ + h) * S_ + s) * HD_ + 2 * d2;
    const float cr = fre[s * 128 + d2];
    const float ci = fim[s * 128 + d2];

    {   // Q: rope + scale
        float re = qkv[src + 2 * d2], im = qkv[src + 2 * d2 + 1];
        qf[dst]     = (_Float16)((re * cr - im * ci) * SCALE_);
        qf[dst + 1] = (_Float16)((re * ci + im * cr) * SCALE_);
    }
    {   // K: rope
        float re = qkv[src + HID_ + 2 * d2], im = qkv[src + HID_ + 2 * d2 + 1];
        kf[dst]     = (_Float16)(re * cr - im * ci);
        kf[dst + 1] = (_Float16)(re * ci + im * cr);
    }
    {   // V: copy
        vf[dst]     = (_Float16)qkv[src + 2 * HID_ + 2 * d2];
        vf[dst + 1] = (_Float16)qkv[src + 2 * HID_ + 2 * d2 + 1];
    }
}

// -------------- V transpose: [B,NH,S,HD] -> [B,NH,HD,S] (f16) --------------
// 64x64 tiles through LDS; coalesced loads and stores.
__global__ void __launch_bounds__(256) transpose_v(const _Float16* __restrict__ v,
                                                   _Float16* __restrict__ vt) {
    __shared__ _Float16 t[64][65];            // +1 pad: no bank conflicts
    const int blk = blockIdx.x;               // B*NH * (S/64) * (HD/64)
    const int td  = blk & (HD_ / 64 - 1);     // 0..3
    const int ts  = (blk >> 2) & (S_ / 64 - 1);
    const int bh  = blk >> 7;

    const _Float16* src = v  + ((size_t)bh * S_  + ts * 64) * HD_ + td * 64;
    _Float16*       dst = vt + ((size_t)bh * HD_ + td * 64) * S_  + ts * 64;

#pragma unroll
    for (int i = 0; i < 16; ++i) {
        const int lin = i * 256 + threadIdx.x;
        t[lin >> 6][lin & 63] = src[(size_t)(lin >> 6) * HD_ + (lin & 63)];
    }
    __syncthreads();
#pragma unroll
    for (int i = 0; i < 16; ++i) {
        const int lin = i * 256 + threadIdx.x;
        dst[(size_t)(lin >> 6) * S_ + (lin & 63)] = t[lin & 63][lin >> 6];
    }
}

// ------------------------- flash attention (causal) ------------------------
// one wave per (b,h, 16-row Q tile); K tiles of 32; online softmax;
// O accumulator = 16 x 256 fp32 held in 16 WMMA C-fragments.
__global__ void __launch_bounds__(256) attn_flash(const _Float16* __restrict__ q,
                                                  const _Float16* __restrict__ k,
                                                  const _Float16* __restrict__ vt,
                                                  _Float16* __restrict__ attn) {
    __shared__ _Float16 lds[8][16 * 32];   // per-wave P-transpose staging

    const int lane = threadIdx.x & 31;
    const int wib  = threadIdx.x >> 5;
    const int wid  = blockIdx.x * 8 + wib;
    const int bh = wid >> 7;               // 0..B*NH-1
    const int qt = wid & 127;              // 0..S/16-1
    const int m0 = qt * 16;
    const int grp = (lane >> 4) & 1;
    const int idx = lane & 15;

    const _Float16* qp  = q  + (size_t)bh * S_ * HD_;
    const _Float16* kp  = k  + (size_t)bh * S_ * HD_;
    const _Float16* vtp = vt + (size_t)bh * HD_ * S_;   // [HD, S]

    // Q fragments for the whole head dim (8 x K=32), register resident
    v16h qa[8];
#pragma unroll
    for (int t = 0; t < 8; ++t) qa[t] = load_frag_row(qp, HD_, m0, t * 32, lane);

    v8f o[16];
#pragma unroll
    for (int j = 0; j < 16; ++j)
#pragma unroll
        for (int r = 0; r < 8; ++r) o[j][r] = 0.0f;

    float mrow[8], lrow[8];
#pragma unroll
    for (int r = 0; r < 8; ++r) { mrow[r] = -3.0e38f; lrow[r] = 0.0f; }

    for (int kb = 0; kb <= m0 + 15; kb += 32) {
        // prefetch next K tile rows (global_prefetch_b8)
        if (kb + 32 <= m0 + 15) {
            __builtin_prefetch(kp + (size_t)(kb + 32 + idx) * HD_, 0, 3);
            __builtin_prefetch(vtp + (size_t)idx * S_ + kb + 32, 0, 3);
        }

        // ---- scores 16x32 : two C-frags, 16 WMMAs ----
        v8f s0 = {}, s1 = {};
#pragma unroll
        for (int t = 0; t < 8; ++t) {
            v16h bk0 = load_frag_row(kp, HD_, kb +  0, t * 32, lane);
            v16h bk1 = load_frag_row(kp, HD_, kb + 16, t * 32, lane);
            s0 = wmma16x16x32(qa[t], bk0, s0);
            s1 = wmma16x16x32(qa[t], bk1, s1);
        }

        // ---- causal mask + online softmax (row m = r + grp*8 + m0) ----
        float alpha[8];
#pragma unroll
        for (int r = 0; r < 8; ++r) {
            const int mg = m0 + r + grp * 8;
            float a0 = (kb + idx      <= mg) ? s0[r] : -1.0e9f;
            float a1 = (kb + 16 + idx <= mg) ? s1[r] : -1.0e9f;
            float mx = fmaxf(a0, a1);
            mx = fmaxf(mx, __shfl_xor(mx, 1, 32));
            mx = fmaxf(mx, __shfl_xor(mx, 2, 32));
            mx = fmaxf(mx, __shfl_xor(mx, 4, 32));
            mx = fmaxf(mx, __shfl_xor(mx, 8, 32));
            const float mnew = fmaxf(mrow[r], mx);
            const float p0 = __expf(a0 - mnew);
            const float p1 = __expf(a1 - mnew);
            alpha[r] = __expf(mrow[r] - mnew);
            float rs = p0 + p1;
            rs += __shfl_xor(rs, 1, 32);
            rs += __shfl_xor(rs, 2, 32);
            rs += __shfl_xor(rs, 4, 32);
            rs += __shfl_xor(rs, 8, 32);
            lrow[r] = lrow[r] * alpha[r] + rs;
            mrow[r] = mnew;
            // stage P (f16) for transpose to A-fragment layout
            lds[wib][(r + grp * 8) * 32 + idx]      = (_Float16)p0;
            lds[wib][(r + grp * 8) * 32 + 16 + idx] = (_Float16)p1;
        }

        // wave-internal LDS RAW: all lanes' ds_stores must land before reads
        asm volatile("s_wait_dscnt 0" ::: "memory");

        v16h pa = load_frag_row(&lds[wib][0], 32, 0, 0, lane);

        // ---- O = diag(alpha)*O + P @ V   (V^T rows = output columns) ----
#pragma unroll
        for (int j = 0; j < 16; ++j) {
#pragma unroll
            for (int r = 0; r < 8; ++r) o[j][r] *= alpha[r];
            v16h bv = load_frag_row(vtp, S_, j * 16, kb, lane);
            o[j] = wmma16x16x32(pa, bv, o[j]);
        }
    }

    // ---- normalize and store to attn_f16 [B,S,HID] at col h*HD ----
    float inv[8];
#pragma unroll
    for (int r = 0; r < 8; ++r) inv[r] = 1.0f / lrow[r];
    const int b = bh >> 2;
    const int h = bh & 3;
#pragma unroll
    for (int j = 0; j < 16; ++j) {
#pragma unroll
        for (int r = 0; r < 8; ++r) {
            const size_t row = (size_t)(b * S_ + m0 + r + grp * 8);
            attn[row * HID_ + h * HD_ + j * 16 + idx] = (_Float16)(o[j][r] * inv[r]);
        }
    }
}

// ---------------------------------------------------------------------------
extern "C" void kernel_launch(void* const* d_in, const int* in_sizes, int n_in,
                              void* d_out, int out_size, void* d_ws, size_t ws_size,
                              hipStream_t stream) {
    (void)in_sizes; (void)n_in; (void)out_size; (void)ws_size;
    const float* hidden = (const float*)d_in[0];
    const float* fre    = (const float*)d_in[1];
    const float* fim    = (const float*)d_in[2];
    /* mask d_in[3] unused: causal handled analytically */
    const float* W_qkv  = (const float*)d_in[4];
    const float* W_o    = (const float*)d_in[5];
    float* out = (float*)d_out;

    const int M = B_ * S_;                      // 4096
    char* ws = (char*)d_ws;
    size_t off = 0;
    _Float16* h16   = (_Float16*)(ws + off); off += (size_t)M * HID_ * 2;          // 8 MB
    _Float16* wq16  = (_Float16*)(ws + off); off += (size_t)3 * HID_ * HID_ * 2;   // 6 MB
    _Float16* wo16  = (_Float16*)(ws + off); off += (size_t)HID_ * HID_ * 2;       // 2 MB
    float*    qkv32 = (float*)   (ws + off); off += (size_t)M * 3 * HID_ * 4;      // 48 MB
    _Float16* q16   = (_Float16*)(ws + off); off += (size_t)M * HID_ * 2;
    _Float16* k16   = (_Float16*)(ws + off); off += (size_t)M * HID_ * 2;
    _Float16* v16   = (_Float16*)(ws + off); off += (size_t)M * HID_ * 2;
    _Float16* vt16  = (_Float16*)(ws + off); off += (size_t)M * HID_ * 2;
    _Float16* a16   = (_Float16*)(ws + off); off += (size_t)M * HID_ * 2;

    // 1) downconvert activations + weights
    cvt_f32_f16<<<(M * HID_) / 256, 256, 0, stream>>>(hidden, h16, M * HID_);
    cvt_f32_f16<<<(3 * HID_ * HID_) / 256, 256, 0, stream>>>(W_qkv, wq16, 3 * HID_ * HID_);
    cvt_f32_f16<<<(HID_ * HID_) / 256, 256, 0, stream>>>(W_o, wo16, HID_ * HID_);

    // 2) QKV projection: [4096,1024] x [3072,1024]^T -> [4096,3072] f32
    {
        int waves = (M / 16) * ((3 * HID_) / 64);       // 12288
        gemm_wmma<<<waves * 32 / 256, 256, 0, stream>>>(h16, wq16, qkv32,
                                                        M, 3 * HID_, HID_);
    }

    // 3) RoPE + scale + repack to per-head f16
    rope_cvt<<<(B_ * S_ * NH_ * (HD_ / 2)) / 256, 256, 0, stream>>>(
        qkv32, fre, fim, q16, k16, v16);

    // 3b) V transpose for vectorized P@V B-fragments
    transpose_v<<<B_ * NH_ * (S_ / 64) * (HD_ / 64), 256, 0, stream>>>(v16, vt16);

    // 4) flash attention: 1 wave per (b,h,q-tile) = 8*128 waves
    attn_flash<<<(B_ * NH_ * (S_ / 16)) * 32 / 256, 256, 0, stream>>>(
        q16, k16, vt16, a16);

    // 5) output projection: [4096,1024] x [1024,1024]^T -> d_out f32
    {
        int waves = (M / 16) * (HID_ / 64);             // 4096
        gemm_wmma<<<waves * 32 / 256, 256, 0, stream>>>(a16, wo16, out,
                                                        M, HID_, HID_);
    }
}
